// VeryModel_8624294331056
// MI455X (gfx1250) — matrix operand order repaired and checked by reading, not compiled, not run on hardware
//
#include <hip/hip_runtime.h>
#include <hip/hip_bf16.h>
#include <math.h>

// ---------------------------------------------------------------------------
// Model dims
//   B=512, CAP=33, T=32, BT=16384
//   out1: 512x120  (K=2048)      naive f32
//   out2: 512x300  (K=120)       naive f32
//   hp  : 512x1200 (K=300)       naive f32   (includes b_hh + b_ih)
//   gates: 16384x1200 (K=100->128)  WMMA bf16, fused LSTM cell -> hb (bf16, Kpad 320)
//   out3: 16384x120 (K=300->320)    WMMA bf16 -> out3b (bf16, Kpad 128)
//   out4: 16384x8769 (K=120->128)   WMMA bf16 -> d_out (f32)
// ---------------------------------------------------------------------------

typedef __attribute__((ext_vector_type(16))) __bf16        v16bf;
typedef __attribute__((ext_vector_type(8)))  float         v8f;
typedef __attribute__((ext_vector_type(4)))  unsigned int  v4u;
typedef __attribute__((ext_vector_type(8)))  unsigned int  v8u;

__device__ __forceinline__ unsigned short f2bf(float f) {
  unsigned int u = __builtin_bit_cast(unsigned int, f);
  u += 0x7FFFu + ((u >> 16) & 1u);              // round-to-nearest-even
  return (unsigned short)(u >> 16);
}
__device__ __forceinline__ float elu1(float x) { return x > 0.f ? x : expm1f(x); }
__device__ __forceinline__ float sigm(float x) { return 1.f / (1.f + __expf(-x)); }

// Load a 16-element bf16 fragment: 8 contiguous bf16 at p, 8 contiguous at p+gap.
// A-frag and B-frag both use gap=16 (K 8h..8h+7 and 16+8h..16+8h+7 per lane-half).
__device__ __forceinline__ v16bf frag_lo_hi(const unsigned short* p, int gap) {
  v4u a0 = *reinterpret_cast<const v4u*>(p);
  v4u a1 = *reinterpret_cast<const v4u*>(p + gap);
  v8u t;
  t[0] = a0[0]; t[1] = a0[1]; t[2] = a0[2]; t[3] = a0[3];
  t[4] = a1[0]; t[5] = a1[1]; t[6] = a1[2]; t[7] = a1[3];
  return __builtin_bit_cast(v16bf, t);
}

__device__ __forceinline__ v8f wmma_bf16(v16bf a, v16bf b, v8f c) {
  return __builtin_amdgcn_wmma_f32_16x16x32_bf16(false, a, false, b, (short)0, c,
                                                 false, false);
}

// ------------------------------ prep kernels -------------------------------

__global__ void k_gt(float* __restrict__ out, const int* __restrict__ cap) {
  int i = blockIdx.x * blockDim.x + threadIdx.x;
  if (i >= 16384) return;
  int b = i >> 5, t = i & 31;
  int g = cap[b * 33 + t + 1];
  out[i] = (g == 1) ? 0.f : (float)g;
}

// dst[rows_alloc][dst_cols] bf16 <- src[rows_valid][src_cols] f32, zero-padded
__global__ void k_convert_pad(unsigned short* __restrict__ dst,
                              const float* __restrict__ src,
                              int rows_alloc, int rows_valid,
                              int src_cols, int dst_cols) {
  size_t total = (size_t)rows_alloc * dst_cols;
  for (size_t i = (size_t)blockIdx.x * blockDim.x + threadIdx.x; i < total;
       i += (size_t)gridDim.x * blockDim.x) {
    int k = (int)(i % dst_cols);
    int n = (int)(i / dst_cols);
    float v = (n < rows_valid && k < src_cols) ? src[(size_t)n * src_cols + k] : 0.f;
    dst[i] = f2bf(v);
  }
}

// xb[16384][128] = bf16(embed[caption[b][t]][k]) (k<100), pad 0
__global__ void k_embed_gather(unsigned short* __restrict__ xb,
                               const int* __restrict__ cap,
                               const float* __restrict__ emb) {
  size_t total = (size_t)16384 * 128;
  for (size_t i = (size_t)blockIdx.x * blockDim.x + threadIdx.x; i < total;
       i += (size_t)gridDim.x * blockDim.x) {
    int k = (int)(i & 127);
    int bt = (int)(i >> 7);
    int b = bt >> 5, t = bt & 31;
    int tok = cap[b * 33 + t];
    float v = (k < 100) ? emb[(size_t)tok * 100 + k] : 0.f;
    xb[i] = f2bf(v);
  }
}

__global__ void k_zero32(unsigned int* __restrict__ p, size_t n32) {
  for (size_t i = (size_t)blockIdx.x * blockDim.x + threadIdx.x; i < n32;
       i += (size_t)gridDim.x * blockDim.x)
    p[i] = 0u;
}

// ------------------------- small f32 VALU GEMMs ----------------------------

__global__ __launch_bounds__(128) void k_out1(const float* __restrict__ img,
                                              const float* __restrict__ W1,
                                              const float* __restrict__ b1,
                                              float* __restrict__ out1) {
  int b = blockIdx.x, n = threadIdx.x;
  if (n >= 120) return;
  const float* x = img + (size_t)b * 2048;
  const float* w = W1 + (size_t)n * 2048;
  float acc = b1[n];
#pragma unroll 8
  for (int k = 0; k < 2048; ++k) acc = fmaf(x[k], w[k], acc);
  out1[b * 120 + n] = elu1(acc);
}

__global__ __launch_bounds__(320) void k_out2(const float* __restrict__ out1,
                                              const float* __restrict__ W2,
                                              const float* __restrict__ b2,
                                              float* __restrict__ out2) {
  int b = blockIdx.x, n = threadIdx.x;
  if (n >= 300) return;
  const float* x = out1 + b * 120;
  const float* w = W2 + (size_t)n * 120;
  float acc = b2[n];
#pragma unroll 8
  for (int k = 0; k < 120; ++k) acc = fmaf(x[k], w[k], acc);
  out2[b * 300 + n] = elu1(acc);
}

// hp = out2 @ W_hh^T + b_hh + b_ih   (fold in b_ih so gates add one term)
__global__ void k_hpart(const float* __restrict__ out2,
                        const float* __restrict__ Whh,
                        const float* __restrict__ bhh,
                        const float* __restrict__ bih,
                        float* __restrict__ hp) {
  int i = blockIdx.x * blockDim.x + threadIdx.x;
  if (i >= 512 * 1200) return;
  int b = i / 1200, n = i % 1200;
  const float* x = out2 + b * 300;
  const float* w = Whh + (size_t)n * 300;
  float acc = bhh[n] + bih[n];
#pragma unroll 4
  for (int k = 0; k < 300; ++k) acc = fmaf(x[k], w[k], acc);
  hp[i] = acc;
}

// --------------------------- WMMA GEMM kernels -----------------------------

// gates (4 x 16384x300, K=128) + fused LSTM cell -> hb[16384][320] bf16
__global__ __launch_bounds__(32) void k_gates(const unsigned short* __restrict__ xb,
                                              const unsigned short* __restrict__ wihb,
                                              const float* __restrict__ hp,
                                              const float* __restrict__ out2,
                                              unsigned short* __restrict__ hb) {
  const int lane = threadIdx.x & 31;
  const int hi = lane >> 4, l15 = lane & 15;
  const int m0 = blockIdx.x * 16;
  const int u0 = blockIdx.y * 16;
  const int uraw = u0 + l15;
  const int uc = uraw < 300 ? uraw : 299;   // clamp B-row loads on last tile

  const unsigned short* arow = xb + (size_t)(m0 + l15) * 128 + hi * 8;
  const unsigned short* bi = wihb + (size_t)(0 * 300 + uc) * 128 + hi * 8;
  const unsigned short* bf = wihb + (size_t)(1 * 300 + uc) * 128 + hi * 8;
  const unsigned short* bg = wihb + (size_t)(2 * 300 + uc) * 128 + hi * 8;
  const unsigned short* bo = wihb + (size_t)(3 * 300 + uc) * 128 + hi * 8;

  v8f di = {}, df = {}, dg = {}, dob = {};
#pragma unroll
  for (int s = 0; s < 4; ++s) {
    v16bf a = frag_lo_hi(arow + 32 * s, 16);
    di  = wmma_bf16(a, frag_lo_hi(bi + 32 * s, 16), di);
    df  = wmma_bf16(a, frag_lo_hi(bf + 32 * s, 16), df);
    dg  = wmma_bf16(a, frag_lo_hi(bg + 32 * s, 16), dg);
    dob = wmma_bf16(a, frag_lo_hi(bo + 32 * s, 16), dob);
  }

  if (uraw < 300) {
#pragma unroll
    for (int r = 0; r < 8; ++r) {
      int m = m0 + r + 8 * hi;          // C layout: vgpr r -> M=r (lo) / r+8 (hi)
      int b = m >> 5;
      float gi = di[r]  + hp[b * 1200 + uraw];
      float gf = df[r]  + hp[b * 1200 + 300 + uraw];
      float gg = dg[r]  + hp[b * 1200 + 600 + uraw];
      float go = dob[r] + hp[b * 1200 + 900 + uraw];
      float c = sigm(gf) * out2[b * 300 + uraw] + sigm(gi) * tanhf(gg);
      float h = sigm(go) * tanhf(c);
      hb[(size_t)m * 320 + uraw] = f2bf(h);
    }
  }
}

// out3 = elu(hb @ W3^T + b3) -> out3b[16384][128] bf16 (cols 120..127 zero)
__global__ __launch_bounds__(32) void k_out3(const unsigned short* __restrict__ hb,
                                             const unsigned short* __restrict__ w3b,
                                             const float* __restrict__ b3,
                                             unsigned short* __restrict__ out3b) {
  const int lane = threadIdx.x & 31;
  const int hi = lane >> 4, l15 = lane & 15;
  const int m0 = blockIdx.x * 16;
  const int n = blockIdx.y * 16 + l15;   // < 128, w3b has 128 rows (120 valid)

  const unsigned short* arow = hb + (size_t)(m0 + l15) * 320 + hi * 8;
  const unsigned short* brow = w3b + (size_t)n * 320 + hi * 8;

  v8f acc = {};
#pragma unroll
  for (int s = 0; s < 10; ++s)
    acc = wmma_bf16(frag_lo_hi(arow + 32 * s, 16), frag_lo_hi(brow + 32 * s, 16), acc);

  float bias = (n < 120) ? b3[n] : 0.f;
#pragma unroll
  for (int r = 0; r < 8; ++r) {
    int m = m0 + r + 8 * hi;
    float v = (n < 120) ? elu1(acc[r] + bias) : 0.f;
    out3b[(size_t)m * 128 + n] = f2bf(v);
  }
}

// out4 = out3 @ W4^T + b4 -> f32, 16384x8769. 8 waves/block, one 16x16 tile/wave.
__global__ __launch_bounds__(256) void k_out4(const unsigned short* __restrict__ out3b,
                                              const unsigned short* __restrict__ w4b,
                                              const float* __restrict__ b4,
                                              float* __restrict__ out) {
  const int lane = threadIdx.x & 31;
  const int wave = threadIdx.x >> 5;
  const int hi = lane >> 4, l15 = lane & 15;
  const int m0 = blockIdx.x * 16;
  const int tn = blockIdx.y * 8 + wave;
  if (tn >= 549) return;                 // wave-uniform: EXEC stays all-ones
  const int n = tn * 16 + l15;           // < 8784, w4b has 8784 rows (8769 valid)

  const unsigned short* arow = out3b + (size_t)(m0 + l15) * 128 + hi * 8;
  const unsigned short* brow = w4b + (size_t)n * 128 + hi * 8;

  v8f acc = {};
#pragma unroll
  for (int s = 0; s < 4; ++s)
    acc = wmma_bf16(frag_lo_hi(arow + 32 * s, 16), frag_lo_hi(brow + 32 * s, 16), acc);

  if (n < 8769) {
    float bias = b4[n];
#pragma unroll
    for (int r = 0; r < 8; ++r) {
      int m = m0 + r + 8 * hi;
      out[(size_t)m * 8769 + n] = acc[r] + bias;
    }
  }
}

// ------------------------------- launcher ----------------------------------

extern "C" void kernel_launch(void* const* d_in, const int* in_sizes, int n_in,
                              void* d_out, int out_size, void* d_ws, size_t ws_size,
                              hipStream_t stream) {
  (void)in_sizes; (void)n_in; (void)out_size; (void)ws_size;
  const float* img = (const float*)d_in[0];
  const int*   cap = (const int*)d_in[1];
  const float* W1  = (const float*)d_in[2];
  const float* b1  = (const float*)d_in[3];
  const float* W2  = (const float*)d_in[4];
  const float* b2  = (const float*)d_in[5];
  const float* emb = (const float*)d_in[6];
  const float* Wih = (const float*)d_in[7];
  const float* Whh = (const float*)d_in[8];
  const float* bih = (const float*)d_in[9];
  const float* bhh = (const float*)d_in[10];
  const float* W3  = (const float*)d_in[11];
  const float* b3  = (const float*)d_in[12];
  const float* W4  = (const float*)d_in[13];
  const float* b4  = (const float*)d_in[14];
  float* out = (float*)d_out;

  char* ws = (char*)d_ws;
  size_t off = 0;
  auto alloc = [&](size_t bytes) -> void* {
    void* p = ws + off;
    off += (bytes + 255) & ~(size_t)255;
    return p;
  };
  float* out1 = (float*)alloc((size_t)512 * 120 * 4);
  float* out2 = (float*)alloc((size_t)512 * 300 * 4);
  float* hp   = (float*)alloc((size_t)512 * 1200 * 4);
  unsigned short* xb    = (unsigned short*)alloc((size_t)16384 * 128 * 2);
  unsigned short* wihb  = (unsigned short*)alloc((size_t)1200 * 128 * 2);
  unsigned short* hb    = (unsigned short*)alloc((size_t)16384 * 320 * 2);
  unsigned short* w3b   = (unsigned short*)alloc((size_t)128 * 320 * 2);
  unsigned short* out3b = (unsigned short*)alloc((size_t)16384 * 128 * 2);
  unsigned short* w4b   = (unsigned short*)alloc((size_t)8784 * 128 * 2);

  // gt output (first 16384 floats of d_out)
  k_gt<<<64, 256, 0, stream>>>(out, cap);

  // weight/activation conversions to padded bf16
  k_convert_pad<<<600, 256, 0, stream>>>(wihb, Wih, 1200, 1200, 100, 128);
  k_convert_pad<<<160, 256, 0, stream>>>(w3b, W3, 128, 120, 300, 320);
  k_convert_pad<<<4392, 256, 0, stream>>>(w4b, W4, 8784, 8769, 120, 128);
  k_embed_gather<<<8192, 256, 0, stream>>>(xb, cap, emb);
  k_zero32<<<10240, 256, 0, stream>>>((unsigned int*)hb,
                                      (size_t)16384 * 320 * 2 / 4);

  // small f32 stages
  k_out1<<<512, 128, 0, stream>>>(img, W1, b1, out1);
  k_out2<<<512, 320, 0, stream>>>(out1, W2, b2, out2);
  k_hpart<<<2400, 256, 0, stream>>>(out2, Whh, bhh, bih, hp);

  // WMMA stages
  k_gates<<<dim3(1024, 19), 32, 0, stream>>>(xb, wihb, hp, out2, hb);
  k_out3<<<dim3(1024, 8), 32, 0, stream>>>(hb, w3b, b3, out3b);
  k_out4<<<dim3(1024, 69), 256, 0, stream>>>(out3b, w4b, b4, out + 16384);
}
